// GNN_LSTM_Gravity_25838523253466
// MI455X (gfx1250) — matrix operand order repaired and checked by reading, not compiled, not run on hardware
//
#include <hip/hip_runtime.h>
#include <math.h>

#define NN 100000
#define NE 640000
#define FD 128
#define HD 128

typedef __attribute__((ext_vector_type(16))) __bf16 v16bf;
typedef __attribute__((ext_vector_type(8)))  __bf16 v8bf;
typedef __attribute__((ext_vector_type(8)))  float  v8f;

union V16U { v16bf v; v8bf h[2]; };

// A-fragment per-lane load: 16 contiguous bf16 at K=k0..k0+15 (two 16B aligned halves)
__device__ __forceinline__ v16bf ldA(const __bf16* p) {
  V16U u;
  u.h[0] = *(const v8bf*)p;
  u.h[1] = *(const v8bf*)(p + 8);
  return u.v;
}
// B-fragment per-lane load: 16 contiguous bf16 (32B aligned); works for global or LDS ptrs
__device__ __forceinline__ v16bf ldB(const __bf16* p) {
  return *(const v16bf*)p;
}
__device__ __forceinline__ float sigm(float x) { return 1.0f / (1.0f + __expf(-x)); }

// ---- gfx1250 async copy: 16 bytes per lane, global -> LDS, tracked by ASYNCcnt ----
__device__ __forceinline__ void async_copy16(unsigned ldsOff, const void* g) {
  asm volatile("global_load_async_to_lds_b128 %0, %1, off"
               :: "v"(ldsOff), "v"((unsigned long long)(uintptr_t)g)
               : "memory");
}
__device__ __forceinline__ void async_wait0() {
  asm volatile("s_wait_asynccnt 0x0" ::: "memory");
}
__device__ __forceinline__ unsigned lds_off(const void* p) {
  // generic pointer to LDS: low 32 bits are the LDS byte address
  return (unsigned)(uintptr_t)p;
}

// ---------- init: deg=1 (self loop), agg=0 ----------
__global__ void k_init(float* deg, float* agg) {
  int i = blockIdx.x * blockDim.x + threadIdx.x;
  if (i < NN * HD) agg[i] = 0.0f;
  if (i < NN) deg[i] = 1.0f;
}

// ---------- fp32 -> bf16 convert ----------
__global__ void k_cvt(const float* __restrict__ src, __bf16* __restrict__ dst, int n) {
  int i = blockIdx.x * blockDim.x + threadIdx.x;
  if (i < n) dst[i] = (__bf16)src[i];
}

// ---------- degree accumulation over dst ----------
__global__ void k_deg(const int* __restrict__ ei, float* deg) {
  int e = blockIdx.x * blockDim.x + threadIdx.x;
  if (e < NE) atomicAdd(&deg[ei[NE + e]], 1.0f);
}

// ---------- dinv = rsqrt(max(deg,eps)), in place ----------
__global__ void k_dinv(float* deg) {
  int i = blockIdx.x * blockDim.x + threadIdx.x;
  if (i < NN) deg[i] = rsqrtf(fmaxf(deg[i], 1e-12f));
}

// ---------- x = nf @ W_gcn.T  (bf16 WMMA, B staged in LDS via async DMA) ----------
__global__ void __launch_bounds__(256)
k_gcn_gemm(const __bf16* __restrict__ nf, const __bf16* __restrict__ wg,
           float* __restrict__ x) {
  __shared__ __bf16 sW[HD * FD];                     // 32 KB: whole W_gcn (bf16)
  const int tid = threadIdx.x;
  // stage: 2048 16B-chunks across 256 threads = 8 async ops per lane
#pragma unroll
  for (int i = 0; i < 8; ++i) {
    int c = tid + i * 256;
    async_copy16(lds_off(&sW[c * 8]), wg + (size_t)c * 8);
  }
  async_wait0();
  __syncthreads();

  int wave = (int)((blockIdx.x * blockDim.x + tid) >> 5);
  int lane = tid & 31;
  int l16 = lane & 15, half = lane >> 4;
  bool valid = (wave < NN / 16);
  int nodeBase = (valid ? wave : 0) * 16;
  const __bf16* rowA = nf + (size_t)(nodeBase + l16) * FD;
  v16bf a[4];
#pragma unroll
  for (int kf = 0; kf < 4; ++kf) a[kf] = ldA(rowA + kf * 32 + half * 16);
  if (!valid) return;                                // no barriers after this point
#pragma unroll 1
  for (int ht = 0; ht < 8; ++ht) {
    v8f acc = {};
    const __bf16* rowB = &sW[(ht * 16 + l16) * FD + half * 16];
#pragma unroll
    for (int kf = 0; kf < 4; ++kf) {
      acc = __builtin_amdgcn_wmma_f32_16x16x32_bf16(false, a[kf], false, ldB(rowB + kf * 32),
                                                    (short)0, acc, false, false);
    }
    float* xp = x + (size_t)(nodeBase + half * 8) * HD + ht * 16 + l16;
#pragma unroll
    for (int r = 0; r < 8; ++r) xp[(size_t)r * HD] = acc[r];
  }
}

// ---------- scatter: agg[dst] += dinv[src]*dinv[dst] * x[src]  (incl. self loops) ----------
__global__ void k_scatter(const int* __restrict__ ei, const float* __restrict__ dinv,
                          const float* __restrict__ x, float* agg) {
  long long gid = (long long)blockIdx.x * blockDim.x + threadIdx.x;
  int e = (int)(gid >> 5);
  int sub = (int)(gid & 31);
  if (e >= NE + NN) return;
  int s, d;
  if (e < NE) { s = ei[e]; d = ei[NE + e]; } else { s = d = e - NE; }
  float nrm = dinv[s] * dinv[d];
  const float4 xv = *(const float4*)(x + (size_t)s * HD + sub * 4);
  float* ap = agg + (size_t)d * HD + sub * 4;
  atomicAdd(ap + 0, nrm * xv.x);
  atomicAdd(ap + 1, nrm * xv.y);
  atomicAdd(ap + 2, nrm * xv.z);
  atomicAdd(ap + 3, nrm * xv.w);
}

// ---------- h = bf16(relu(agg + b_gcn)) ----------
__global__ void k_relu(const float* __restrict__ agg, const float* __restrict__ b_gcn,
                       __bf16* __restrict__ h) {
  int i = blockIdx.x * blockDim.x + threadIdx.x;
  if (i >= NN * HD) return;
  float v = agg[i] + b_gcn[i & (HD - 1)];
  h[i] = (__bf16)fmaxf(v, 0.0f);
}

// ---------- fused: gather ef -> WMMA gates (i,g,o; f unused since c0=0) -> LSTM -> dot W_fc
// W_ih slices double-buffered in LDS via async DMA, overlapped with WMMA.
__global__ void __launch_bounds__(256)
k_edge_lstm(const int* __restrict__ ei, const __bf16* __restrict__ h,
            const __bf16* __restrict__ wih, const float* __restrict__ b_ih,
            const float* __restrict__ b_hh, const float* __restrict__ w_fc,
            const float* __restrict__ b_fc, float* __restrict__ out) {
  // per stage: 3 gates x 16 cols x 256 K bf16 = 24 KB; double buffered = 48 KB
  __shared__ __bf16 sB[2][3 * 16 * 256];
  const int tid = threadIdx.x;
  int wave = (int)((blockIdx.x * blockDim.x + tid) >> 5);
  int lane = tid & 31;
  int l16 = lane & 15, half = lane >> 4;
  bool valid = (wave < NE / 16);
  int eBase = (valid ? wave : 0) * 16;
  int e = eBase + l16;                 // lanes l and l+16 handle the same edge row
  int s = ei[e], d = ei[NE + e];
  const __bf16* rowS = h + (size_t)s * HD;
  const __bf16* rowD = h + (size_t)d * HD;
  // ef row = [h[src](K=0..127) | h[dst](K=128..255)]; fragments never straddle the concat
  v16bf a[8];
#pragma unroll
  for (int kf = 0; kf < 4; ++kf) {
    a[kf]     = ldA(rowS + kf * 32 + half * 16);
    a[kf + 4] = ldA(rowD + kf * 32 + half * 16);
  }
  float dot[8];
#pragma unroll
  for (int r = 0; r < 8; ++r) dot[r] = 0.0f;
  float bfc = b_fc[0];

  // stage(ct,buf): 1536 16B-chunks over 256 threads = 6 async ops/lane.
  // chunk c: gate=c>>9, within-slice element offset (c&511)*8; LDS is contiguous c*8.
  auto stage = [&](int ct, int buf) {
#pragma unroll
    for (int i = 0; i < 6; ++i) {
      int c = tid + i * 256;
      int gate = c >> 9;
      int gbase = (gate == 0) ? 0 : ((gate == 1) ? 256 * 256 : 384 * 256); // i,g,o row bases
      const __bf16* g = wih + gbase + ct * (16 * 256) + (c & 511) * 8;
      async_copy16(lds_off(&sB[buf][c * 8]), g);
    }
  };

  stage(0, 0);
  async_wait0();
  __syncthreads();

#pragma unroll 1
  for (int ct = 0; ct < 8; ++ct) {
    int cur = ct & 1;
    if (ct < 7) stage(ct + 1, 1 - cur);              // overlap DMA with WMMA below
    int colI = ct * 16 + l16;                        // i gate: W_ih rows [0,128)
    int colG = 256 + colI;                           // g gate: rows [256,384)
    int colO = 384 + colI;                           // o gate: rows [384,512)
    v8f ai = {}, ag = {}, ao = {};
    const __bf16* bI = &sB[cur][(0 * 16 + l16) * 256 + half * 16];
    const __bf16* bG = &sB[cur][(1 * 16 + l16) * 256 + half * 16];
    const __bf16* bO = &sB[cur][(2 * 16 + l16) * 256 + half * 16];
#pragma unroll
    for (int kf = 0; kf < 8; ++kf) {
      ai = __builtin_amdgcn_wmma_f32_16x16x32_bf16(false, a[kf], false, ldB(bI + kf * 32),
                                                   (short)0, ai, false, false);
      ag = __builtin_amdgcn_wmma_f32_16x16x32_bf16(false, a[kf], false, ldB(bG + kf * 32),
                                                   (short)0, ag, false, false);
      ao = __builtin_amdgcn_wmma_f32_16x16x32_bf16(false, a[kf], false, ldB(bO + kf * 32),
                                                   (short)0, ao, false, false);
    }
    float biasI = b_ih[colI] + b_hh[colI];
    float biasG = b_ih[colG] + b_hh[colG];
    float biasO = b_ih[colO] + b_hh[colO];
    float wfc = w_fc[colI];
#pragma unroll
    for (int r = 0; r < 8; ++r) {
      float ig = sigm(ai[r] + biasI);
      float gg = tanhf(ag[r] + biasG);
      float og = sigm(ao[r] + biasO);
      float cs = ig * gg;
      float hs = og * tanhf(cs);
      dot[r] += hs * wfc;                            // column N = l16 of W_fc
    }
    if (ct < 7) {
      async_wait0();                                 // stage(ct+1) landed in LDS
      __syncthreads();                               // visible to all waves; buf reusable
    }
  }
  // reduce over 16 columns inside each half-wave; rows: lanes0-15 -> M=r, lanes16-31 -> M=8+r
#pragma unroll
  for (int r = 0; r < 8; ++r) {
    float v = dot[r];
    v += __shfl_xor(v, 1, 32);
    v += __shfl_xor(v, 2, 32);
    v += __shfl_xor(v, 4, 32);
    v += __shfl_xor(v, 8, 32);
    if (valid && l16 == r) out[eBase + half * 8 + r] = v + bfc;
  }
}

extern "C" void kernel_launch(void* const* d_in, const int* in_sizes, int n_in,
                              void* d_out, int out_size, void* d_ws, size_t ws_size,
                              hipStream_t stream) {
  const float* nf    = (const float*)d_in[0];
  const float* W_gcn = (const float*)d_in[1];
  const float* b_gcn = (const float*)d_in[2];
  const float* W_ih  = (const float*)d_in[3];
  // d_in[4] = W_hh: unused, h0 == 0
  const float* b_ih  = (const float*)d_in[5];
  const float* b_hh  = (const float*)d_in[6];
  const float* W_fc  = (const float*)d_in[7];
  const float* b_fc  = (const float*)d_in[8];
  const int*   ei    = (const int*)d_in[9];
  float* out = (float*)d_out;

  char* ws = (char*)d_ws;
  size_t off = 0;
  auto alloc = [&](size_t bytes) -> void* {
    void* p = ws + off;
    off += bytes;
    off = (off + 255) & ~(size_t)255;
    return p;
  };
  float*  deg  = (float*) alloc((size_t)NN * 4);            // becomes dinv in place
  __bf16* nfb  = (__bf16*)alloc((size_t)NN * FD * 2);
  __bf16* wgb  = (__bf16*)alloc((size_t)HD * FD * 2);
  __bf16* wihb = (__bf16*)alloc((size_t)512 * 256 * 2);
  float*  x    = (float*) alloc((size_t)NN * HD * 4);
  float*  agg  = (float*) alloc((size_t)NN * HD * 4);
  __bf16* hb   = (__bf16*)alloc((size_t)NN * HD * 2);
  (void)ws_size; (void)in_sizes; (void)n_in; (void)out_size;

  const int T = 256;
  k_init<<<(NN * HD + T - 1) / T, T, 0, stream>>>(deg, agg);
  k_cvt<<<(NN * FD + T - 1) / T, T, 0, stream>>>(nf, nfb, NN * FD);
  k_cvt<<<(HD * FD + T - 1) / T, T, 0, stream>>>(W_gcn, wgb, HD * FD);
  k_cvt<<<(512 * 256 + T - 1) / T, T, 0, stream>>>(W_ih, wihb, 512 * 256);
  k_deg<<<(NE + T - 1) / T, T, 0, stream>>>(ei, deg);
  k_dinv<<<(NN + T - 1) / T, T, 0, stream>>>(deg);
  {
    int waves = NN / 16;                                    // 6250
    k_gcn_gemm<<<(waves * 32 + T - 1) / T, T, 0, stream>>>(nfb, wgb, x);
  }
  {
    long long work = (long long)(NE + NN) * 32;
    k_scatter<<<(int)((work + T - 1) / T), T, 0, stream>>>(ei, deg, x, agg);
  }
  k_relu<<<(NN * HD + T - 1) / T, T, 0, stream>>>(agg, b_gcn, hb);
  {
    int waves = NE / 16;                                    // 40000
    k_edge_lstm<<<(waves * 32 + T - 1) / T, T, 0, stream>>>(ei, hb, wihb, b_ih, b_hh,
                                                            W_fc, b_fc, out);
  }
}
